// DeformableTransformerEncoder_33921651704053
// MI455X (gfx1250) — matrix operand order repaired and checked by reading, not compiled, not run on hardware
//
#include <hip/hip_runtime.h>
#include <hip/hip_bf16.h>

#define D_      256
#define NH_     8
#define DH_     32
#define NL_     4
#define NP_     4
#define BATCH   2
#define LQ_     10105
#define MROWS   (BATCH*LQ_)
#define FFN_    1024
#define NLAYERS 6
#define KC_     256              // K staging chunk
#define BPAD_   8                // LDS pad (bf16 elems) -> 16B bank skew
#define BROW_   (KC_ + BPAD_)    // LDS col pitch in elems (528B)
#define PANEL_  (64 * BROW_)     // one staged weight panel (elems)

typedef __attribute__((ext_vector_type(16))) __bf16 v16bf;
typedef __attribute__((ext_vector_type(8)))  __bf16 v8bf;
typedef __attribute__((ext_vector_type(8)))  float  v8f;

// LDS byte offset of a generic pointer to a __shared__ object (ptrtoint of AS3)
__device__ __forceinline__ unsigned lds_off_u32(const void* p) {
    return (unsigned)(unsigned long long)(const __attribute__((address_space(3))) char*)p;
}

// ---------------------------------------------------------------------------
// Flatten pyramid levels + level embedding -> X (M x D), POS (M x D)
// ---------------------------------------------------------------------------
__global__ void flatten_kernel(const float* __restrict__ s0, const float* __restrict__ p0,
                               const float* __restrict__ s1, const float* __restrict__ p1,
                               const float* __restrict__ s2, const float* __restrict__ p2,
                               const float* __restrict__ s3, const float* __restrict__ p3,
                               const float* __restrict__ lvl,
                               float* __restrict__ X, float* __restrict__ POS) {
    int tid = blockIdx.x * blockDim.x + threadIdx.x;
    const int total = MROWS * D_;
    if (tid >= total) return;
    int d   = tid & (D_ - 1);
    int row = tid >> 8;
    int b = row / LQ_;
    int s = row - b * LQ_;
    int l, j, HW;
    const float *sp, *pp;
    if (s < 7600)      { l = 0; j = s;        HW = 7600; sp = s0; pp = p0; }
    else if (s < 9500) { l = 1; j = s - 7600; HW = 1900; sp = s1; pp = p1; }
    else if (s < 9975) { l = 2; j = s - 9500; HW = 475;  sp = s2; pp = p2; }
    else               { l = 3; j = s - 9975; HW = 130;  sp = s3; pp = p3; }
    size_t src_idx = ((size_t)(b * D_ + d)) * HW + j;
    X[tid]   = sp[src_idx] + lvl[l * D_ + d];
    POS[tid] = pp[src_idx];
}

// ---------------------------------------------------------------------------
// Reference points: REF[s*2+0]=x, REF[s*2+1]=y
// ---------------------------------------------------------------------------
__global__ void ref_points_kernel(float* __restrict__ REF) {
    int s = blockIdx.x * blockDim.x + threadIdx.x;
    if (s >= LQ_) return;
    int H, W, j;
    if (s < 7600)      { H = 76; W = 100; j = s; }
    else if (s < 9500) { H = 38; W = 50;  j = s - 7600; }
    else if (s < 9975) { H = 19; W = 25;  j = s - 9500; }
    else               { H = 10; W = 13;  j = s - 9975; }
    int iy = j / W, ix = j - iy * W;
    REF[2 * s + 0] = ((float)ix + 0.5f) / (float)W;
    REF[2 * s + 1] = ((float)iy + 0.5f) / (float)H;
}

// ---------------------------------------------------------------------------
// O = bf16(A [+ B]) elementwise
// ---------------------------------------------------------------------------
__global__ void cast_add_bf16_kernel(const float* __restrict__ A, const float* __restrict__ Bp,
                                     __bf16* __restrict__ O, int n) {
    int i = blockIdx.x * blockDim.x + threadIdx.x;
    if (i >= n) return;
    float v = A[i];
    if (Bp) v += Bp[i];
    O[i] = (__bf16)v;
}

// Transposed weight cast: W (K x N) fp32 -> Wt (N x K) bf16
__global__ void wt_cast_kernel(const float* __restrict__ W, __bf16* __restrict__ Wt,
                               int K, int N) {
    int i = blockIdx.x * blockDim.x + threadIdx.x;
    if (i >= K * N) return;
    int k = i / N, n = i - k * N;
    Wt[(size_t)n * K + k] = (__bf16)W[i];
}

__global__ void copy_kernel(const float* __restrict__ A, float* __restrict__ O, int n) {
    int i = blockIdx.x * blockDim.x + threadIdx.x;
    if (i < n) O[i] = A[i];
}

// ---------------------------------------------------------------------------
// bf16 WMMA GEMM, double-buffered async LDS weight panels:
//   C(MxN) = A(MxK,bf16) @ Wt(NxK,bf16)^T + bias
// Block = 8 waves -> 128x64 macro tile; wave w owns rows [w*16, w*16+16).
// Panel for chunk c+1 is DMA'd (global_load_async_to_lds_b128, ASYNCcnt)
// while chunk c is consumed; 4 B-frags are preloaded per k-step so the 4
// independent WMMA chains issue back-to-back.
// ---------------------------------------------------------------------------
__global__ __launch_bounds__(256)
void gemm_bf16_kernel(const __bf16* __restrict__ A, const __bf16* __restrict__ Wt,
                      const float* __restrict__ bias,
                      float* __restrict__ Cf, __bf16* __restrict__ Cb,
                      int M, int N, int K, int relu) {
    __shared__ __bf16 Bs[2 * PANEL_];           // 2 x 33,792 B

    const int lane = threadIdx.x & 31;
    const int wv   = threadIdx.x >> 5;
    const int nBlocks = N >> 6;                 // 64-col panels
    const int mB = blockIdx.x / nBlocks;
    const int nB = blockIdx.x - mB * nBlocks;
    const int mBase = mB * 128;
    const int nBase = nB * 64;

    const int l16  = lane & 15;
    int rowA = mBase + wv * 16 + l16;
    if (rowA > M - 1) rowA = M - 1;             // clamp; padded rows discarded
    const int aSel = (lane >> 4) ? 8 : 0;       // A-frag K offset (ISA 16-bit A 16x32)
    const int bSel = (lane >> 4) ? 8 : 0;       // B-frag K offset (ISA 16-bit B 32x16)

    v8f acc[4];
#pragma unroll
    for (int nt = 0; nt < 4; ++nt) {
        float bb = bias ? bias[nBase + nt * 16 + l16] : 0.0f;
#pragma unroll
        for (int v = 0; v < 8; ++v) acc[nt][v] = bb;
    }

    const unsigned BsBase = lds_off_u32(Bs);
    const int t  = threadIdx.x;
    const int sr = t >> 2;                      // staging: panel row 0..63
    const int sq = t & 3;                       // staging: 64-elem quarter

    // issue async DMA of one 64x256 weight panel into buffer `buf`
    auto stage = [&](int kc, int buf) {
        const __bf16* src = Wt + (size_t)(nBase + sr) * K + kc + sq * 64;
        unsigned dst = BsBase + (unsigned)(buf * PANEL_ * 2)
                              + (unsigned)((sr * BROW_ + sq * 64) * 2);
#pragma unroll
        for (int i = 0; i < 4; ++i) {
            unsigned long long ga = (unsigned long long)(const void*)(src + i * 8);
            unsigned la = dst + i * 16;
            asm volatile("global_load_async_to_lds_b128 %0, %1, off"
                         :: "v"(la), "v"(ga) : "memory");
        }
    };

    const int nc = K / KC_;
    stage(0, 0);

    for (int c = 0; c < nc; ++c) {
        asm volatile("s_wait_asynccnt 0x0" ::: "memory");
        __syncthreads();                         // panel c visible to all waves;
                                                 // everyone done reading panel c-1
        if (c + 1 < nc) stage((c + 1) * KC_, (c + 1) & 1);

        const __bf16* aRow  = A + (size_t)rowA * K + c * KC_;
        if (c + 1 < nc) __builtin_prefetch(aRow + KC_, 0, 1);
        const __bf16* bPane = Bs + (c & 1) * PANEL_;

        for (int k0 = 0; k0 < KC_; k0 += 32) {
            v8bf a_lo = *(const v8bf*)(aRow + k0 + aSel);
            v8bf a_hi = *(const v8bf*)(aRow + k0 + aSel + 16);
            v16bf af;
#pragma unroll
            for (int j = 0; j < 8; ++j) { af[j] = a_lo[j]; af[j + 8] = a_hi[j]; }

            v16bf bfr[4];
#pragma unroll
            for (int nt = 0; nt < 4; ++nt) {     // preload all 4 B-frags (clause)
                const __bf16* bp = bPane + (nt * 16 + l16) * BROW_ + k0 + bSel;
                v8bf b_lo = *(const v8bf*)bp;
                v8bf b_hi = *(const v8bf*)(bp + 16);
#pragma unroll
                for (int j = 0; j < 8; ++j) { bfr[nt][j] = b_lo[j]; bfr[nt][j + 8] = b_hi[j]; }
            }
#pragma unroll
            for (int nt = 0; nt < 4; ++nt)       // 4 independent WMMA chains
                acc[nt] = __builtin_amdgcn_wmma_f32_16x16x32_bf16(
                              false, af, false, bfr[nt], (short)0, acc[nt], false, false);
        }
    }

    // ---- epilogue -------------------------------------------------------
    const int mB0 = mBase + wv * 16 + ((lane >> 4) ? 8 : 0);
#pragma unroll
    for (int nt = 0; nt < 4; ++nt) {
        const int colN = nBase + nt * 16 + l16;
#pragma unroll
        for (int v = 0; v < 8; ++v) {
            int m = mB0 + v;
            if (m < M) {
                float x = acc[nt][v];
                if (relu) x = x > 0.0f ? x : 0.0f;
                if (Cf) Cf[(size_t)m * N + colN] = x;
                if (Cb) Cb[(size_t)m * N + colN] = (__bf16)x;
            }
        }
    }
}

// ---------------------------------------------------------------------------
// Softmax over the 16 (NL*NP) logits per (row, head), in place
// ---------------------------------------------------------------------------
__global__ void softmax16_kernel(float* __restrict__ AWL, int total) {
    int t = blockIdx.x * blockDim.x + threadIdx.x;
    if (t >= total) return;
    int m = t >> 3, h = t & 7;
    float* p = AWL + (size_t)m * 128 + h * 16;
    float mx = -1e30f;
#pragma unroll
    for (int i = 0; i < 16; ++i) mx = fmaxf(mx, p[i]);
    float s = 0.0f;
    float e[16];
#pragma unroll
    for (int i = 0; i < 16; ++i) { e[i] = __expf(p[i] - mx); s += e[i]; }
    float inv = 1.0f / s;
#pragma unroll
    for (int i = 0; i < 16; ++i) p[i] = e[i] * inv;
}

// ---------------------------------------------------------------------------
// Deformable bilinear sampling: one wave per (row m, head h), lane = channel
// ---------------------------------------------------------------------------
__global__ __launch_bounds__(256)
void deform_kernel(const float* __restrict__ VAL, const float* __restrict__ OFF,
                   const float* __restrict__ AW, const float* __restrict__ REF,
                   float* __restrict__ ATT) {
    int wid  = blockIdx.x * (blockDim.x >> 5) + (threadIdx.x >> 5);
    int lane = threadIdx.x & 31;
    if (wid >= MROWS * NH_) return;
    int m = wid >> 3, h = wid & 7;
    int b = m / LQ_, q = m - b * LQ_;

    const float rx = REF[2 * q], ry = REF[2 * q + 1];
    const float* off = OFF + (size_t)m * 256 + h * 32;
    const float* aw  = AW  + (size_t)m * 128 + h * 16;
    const float* vbase = VAL + (size_t)b * LQ_ * D_ + h * DH_ + lane;

    const int Hs[4] = {76, 38, 19, 10};
    const int Ws[4] = {100, 50, 25, 13};
    const int ST[4] = {0, 7600, 9500, 9975};

    float acc = 0.0f;
#pragma unroll
    for (int l = 0; l < NL_; ++l) {
        const int H = Hs[l], W = Ws[l], st = ST[l];
        const float fW = (float)W, fH = (float)H;
#pragma unroll
        for (int p = 0; p < NP_; ++p) {
            float ox = off[(l * NP_ + p) * 2 + 0];
            float oy = off[(l * NP_ + p) * 2 + 1];
            float px = (rx + ox / fW) * fW - 0.5f;
            float py = (ry + oy / fH) * fH - 0.5f;
            float x0f = floorf(px), y0f = floorf(py);
            float lx = px - x0f, ly = py - y0f;
            int x0 = (int)x0f, y0 = (int)y0f;
            float a = aw[l * NP_ + p];
#pragma unroll
            for (int dy = 0; dy < 2; ++dy) {
#pragma unroll
                for (int dx = 0; dx < 2; ++dx) {
                    int xi = x0 + dx, yi = y0 + dy;
                    bool valid = (xi >= 0) & (xi < W) & (yi >= 0) & (yi < H);
                    int xc = xi < 0 ? 0 : (xi > W - 1 ? W - 1 : xi);
                    int yc = yi < 0 ? 0 : (yi > H - 1 ? H - 1 : yi);
                    float w = (dy ? ly : 1.0f - ly) * (dx ? lx : 1.0f - lx);
                    float g = vbase[(size_t)(st + yc * W + xc) * D_];
                    acc += a * w * (valid ? 1.0f : 0.0f) * g;
                }
            }
        }
    }
    ATT[(size_t)m * D_ + h * DH_ + lane] = acc;
}

// ---------------------------------------------------------------------------
// X = LayerNorm(X + R) * g + b  (one wave per row; lane owns 8 channels)
// ---------------------------------------------------------------------------
__global__ __launch_bounds__(256)
void res_ln_kernel(float* __restrict__ X, const float* __restrict__ R,
                   const float* __restrict__ g, const float* __restrict__ b) {
    int wid  = blockIdx.x * (blockDim.x >> 5) + (threadIdx.x >> 5);
    int lane = threadIdx.x & 31;
    if (wid >= MROWS) return;
    float* xr       = X + (size_t)wid * D_;
    const float* rr = R + (size_t)wid * D_;
    float y[8];
    float s = 0.0f;
#pragma unroll
    for (int j = 0; j < 8; ++j) {
        int c = lane * 8 + j;
        y[j] = xr[c] + rr[c];
        s += y[j];
    }
#pragma unroll
    for (int o = 16; o > 0; o >>= 1) s += __shfl_xor(s, o, 32);
    float mean = s * (1.0f / 256.0f);
    float vs = 0.0f;
#pragma unroll
    for (int j = 0; j < 8; ++j) { float d = y[j] - mean; vs += d * d; }
#pragma unroll
    for (int o = 16; o > 0; o >>= 1) vs += __shfl_xor(vs, o, 32);
    float inv = rsqrtf(vs * (1.0f / 256.0f) + 1e-5f);
#pragma unroll
    for (int j = 0; j < 8; ++j) {
        int c = lane * 8 + j;
        xr[c] = (y[j] - mean) * inv * g[c] + b[c];
    }
}

// ---------------------------------------------------------------------------
extern "C" void kernel_launch(void* const* d_in, const int* in_sizes, int n_in,
                              void* d_out, int out_size, void* d_ws, size_t ws_size,
                              hipStream_t stream) {
    (void)in_sizes; (void)n_in; (void)out_size; (void)ws_size;
    const float* src[4] = {(const float*)d_in[0], (const float*)d_in[2],
                           (const float*)d_in[4], (const float*)d_in[6]};
    const float* pos[4] = {(const float*)d_in[1], (const float*)d_in[3],
                           (const float*)d_in[5], (const float*)d_in[7]};
    const float* lvl    = (const float*)d_in[8];
    const float* w_off  = (const float*)d_in[9];
    const float* b_off  = (const float*)d_in[10];
    const float* w_attn = (const float*)d_in[11];
    const float* b_attn = (const float*)d_in[12];
    const float* w_val  = (const float*)d_in[13];
    const float* b_val  = (const float*)d_in[14];
    const float* w_out  = (const float*)d_in[15];
    const float* b_out  = (const float*)d_in[16];
    const float* ln1_g  = (const float*)d_in[17];
    const float* ln1_b  = (const float*)d_in[18];
    const float* w_f1   = (const float*)d_in[19];
    const float* b_f1   = (const float*)d_in[20];
    const float* w_f2   = (const float*)d_in[21];
    const float* b_f2   = (const float*)d_in[22];
    const float* ln2_g  = (const float*)d_in[23];
    const float* ln2_b  = (const float*)d_in[24];

    // ---- workspace bump allocator -------------------------------------
    char* ws = (char*)d_ws;
    size_t cur = 0;
    auto alloc = [&](size_t bytes) -> void* {
        void* p = ws + cur;
        cur += (bytes + 255) & ~(size_t)255;
        return p;
    };
    const int M = MROWS;
    float* X    = (float*)alloc((size_t)M * D_ * 4);
    float* POS  = (float*)alloc((size_t)M * D_ * 4);
    float* TMP  = (float*)alloc((size_t)M * D_ * 4);
    float* VAL  = (float*)alloc((size_t)M * D_ * 4);
    float* OFFb = (float*)alloc((size_t)M * D_ * 4);
    float* ATT  = (float*)alloc((size_t)M * D_ * 4);
    float* AWL  = (float*)alloc((size_t)M * 128 * 4);
    float* REF  = (float*)alloc((size_t)LQ_ * 2 * 4);
    __bf16* Qb  = (__bf16*)alloc((size_t)M * D_ * 2);
    __bf16* Xb  = (__bf16*)alloc((size_t)M * D_ * 2);
    __bf16* Hb  = (__bf16*)alloc((size_t)M * FFN_ * 2);
    __bf16* wtv = (__bf16*)alloc((size_t)NLAYERS * D_ * D_ * 2);
    __bf16* wto = (__bf16*)alloc((size_t)NLAYERS * D_ * D_ * 2);
    __bf16* wta = (__bf16*)alloc((size_t)NLAYERS * D_ * 128 * 2);
    __bf16* wtu = (__bf16*)alloc((size_t)NLAYERS * D_ * D_ * 2);
    __bf16* wt1 = (__bf16*)alloc((size_t)NLAYERS * D_ * FFN_ * 2);
    __bf16* wt2 = (__bf16*)alloc((size_t)NLAYERS * FFN_ * D_ * 2);

    auto blocks_for = [](long n, int t) { return (int)((n + t - 1) / t); };

    // ---- weight cast + transpose (deterministic, every call) ----------
    for (int l = 0; l < NLAYERS; ++l) {
        wt_cast_kernel<<<blocks_for(D_ * D_, 256), 256, 0, stream>>>(
            w_val + (size_t)l * D_ * D_, wtv + (size_t)l * D_ * D_, D_, D_);
        wt_cast_kernel<<<blocks_for(D_ * D_, 256), 256, 0, stream>>>(
            w_off + (size_t)l * D_ * D_, wto + (size_t)l * D_ * D_, D_, D_);
        wt_cast_kernel<<<blocks_for(D_ * 128, 256), 256, 0, stream>>>(
            w_attn + (size_t)l * D_ * 128, wta + (size_t)l * D_ * 128, D_, 128);
        wt_cast_kernel<<<blocks_for(D_ * D_, 256), 256, 0, stream>>>(
            w_out + (size_t)l * D_ * D_, wtu + (size_t)l * D_ * D_, D_, D_);
        wt_cast_kernel<<<blocks_for(D_ * FFN_, 256), 256, 0, stream>>>(
            w_f1 + (size_t)l * D_ * FFN_, wt1 + (size_t)l * D_ * FFN_, D_, FFN_);
        wt_cast_kernel<<<blocks_for(FFN_ * D_, 256), 256, 0, stream>>>(
            w_f2 + (size_t)l * FFN_ * D_, wt2 + (size_t)l * FFN_ * D_, FFN_, D_);
    }

    // ---- flatten inputs + reference points ----------------------------
    flatten_kernel<<<blocks_for((long)M * D_, 256), 256, 0, stream>>>(
        src[0], pos[0], src[1], pos[1], src[2], pos[2], src[3], pos[3], lvl, X, POS);
    ref_points_kernel<<<blocks_for(LQ_, 256), 256, 0, stream>>>(REF);

    auto gemm = [&](const __bf16* A, const __bf16* Wt, const float* bias,
                    float* Cf, __bf16* Cb, int Mm, int Nn, int Kk, int relu) {
        int blocks = ((Mm + 127) / 128) * (Nn / 64);
        gemm_bf16_kernel<<<blocks, 256, 0, stream>>>(A, Wt, bias, Cf, Cb, Mm, Nn, Kk, relu);
    };

    const int nEl = M * D_;
    for (int l = 0; l < NLAYERS; ++l) {
        // q = x + pos ; value = x @ w_val + b
        cast_add_bf16_kernel<<<blocks_for(nEl, 256), 256, 0, stream>>>(X, POS, Qb, nEl);
        cast_add_bf16_kernel<<<blocks_for(nEl, 256), 256, 0, stream>>>(X, nullptr, Xb, nEl);
        gemm(Xb, wtv + (size_t)l * D_ * D_, b_val + l * D_, VAL, nullptr, M, D_, D_, 0);
        gemm(Qb, wto + (size_t)l * D_ * D_, b_off + l * D_, OFFb, nullptr, M, D_, D_, 0);
        gemm(Qb, wta + (size_t)l * D_ * 128, b_attn + l * 128, AWL, nullptr, M, 128, D_, 0);
        softmax16_kernel<<<blocks_for(M * NH_, 256), 256, 0, stream>>>(AWL, M * NH_);
        deform_kernel<<<blocks_for((long)M * NH_, 8), 256, 0, stream>>>(VAL, OFFb, AWL, REF, ATT);
        // out projection + residual LN1
        cast_add_bf16_kernel<<<blocks_for(nEl, 256), 256, 0, stream>>>(ATT, nullptr, Qb, nEl);
        gemm(Qb, wtu + (size_t)l * D_ * D_, b_out + l * D_, TMP, nullptr, M, D_, D_, 0);
        res_ln_kernel<<<blocks_for(M, 8), 256, 0, stream>>>(X, TMP, ln1_g + l * D_, ln1_b + l * D_);
        // FFN
        cast_add_bf16_kernel<<<blocks_for(nEl, 256), 256, 0, stream>>>(X, nullptr, Xb, nEl);
        gemm(Xb, wt1 + (size_t)l * D_ * FFN_, b_f1 + l * FFN_, nullptr, Hb, M, FFN_, D_, 1);
        gemm(Hb, wt2 + (size_t)l * FFN_ * D_, b_f2 + l * D_, TMP, nullptr, M, D_, FFN_, 0);
        res_ln_kernel<<<blocks_for(M, 8), 256, 0, stream>>>(X, TMP, ln2_g + l * D_, ln2_b + l * D_);
    }

    copy_kernel<<<blocks_for(nEl, 256), 256, 0, stream>>>(X, (float*)d_out, nEl);
}
